// DotGATHead_64321430224948
// MI455X (gfx1250) — matrix operand order repaired and checked by reading, not compiled, not run on hardware
//
#include <hip/hip_runtime.h>
#include <math.h>

// ---------------------------------------------------------------------------
// Types for CDNA5 WMMA (wave32): D(16x16 f32) = A(16x32 f16) x B(32x16 f16) + C
// ---------------------------------------------------------------------------
typedef __attribute__((ext_vector_type(16))) _Float16 v16h;
typedef __attribute__((ext_vector_type(8)))  _Float16 v8h;
typedef __attribute__((ext_vector_type(8)))  float    v8f;
typedef __attribute__((ext_vector_type(4)))  unsigned int u32x4;
typedef __attribute__((ext_vector_type(8)))  int          i32x8;
typedef __attribute__((ext_vector_type(4)))  int          i32x4;

union HV { v16h v; v8h h[2]; _Float16 e[16]; };

#define WMMA_F32_F16(a, b, c) \
    __builtin_amdgcn_wmma_f32_16x16x32_f16(false, (a), false, (b), (short)0, (c), false, false)

static constexpr int   Bb   = 4;
static constexpr int   Aa   = 2048;
static constexpr int   Dd   = 512;
static constexpr int   Hh   = 4;
static constexpr int   HDd  = 128;
static constexpr int   Mm   = Bb * Aa;        // 8192 rows
static constexpr float EPSf = 1e-5f;
static constexpr float SCALEf = 0.08838834764831845f; // 1/sqrt(128)

#if defined(__has_builtin)
#if __has_builtin(__builtin_amdgcn_tensor_load_to_lds)
#define USE_TDM 1
#endif
#endif
#ifndef USE_TDM
#define USE_TDM 0
#endif

// ---------------------------------------------------------------------------
// Tensor Data Mover: 2D tile (2-byte elems) global -> LDS.
// D# layout per CDNA5 ISA.
// ---------------------------------------------------------------------------
#if USE_TDM
__device__ __forceinline__ void tdm_load_2d_f16(void* lds_dst, const void* gsrc,
                                                int tile_w, int tile_h,
                                                int row_stride_elems) {
    const unsigned lds_addr = (unsigned)(size_t)lds_dst;          // low 32b = LDS byte offset
    const unsigned long long ga = (unsigned long long)(size_t)gsrc;
    u32x4 g0;
    g0[0] = 1u;                                                    // count=1, user D#
    g0[1] = lds_addr;                                              // lds_addr
    g0[2] = (unsigned)(ga & 0xFFFFFFFFu);                          // global_addr[31:0]
    g0[3] = (unsigned)((ga >> 32) & 0x01FFFFFFu) | 0x80000000u;    // global_addr[56:32] | type=2
    i32x8 g1;
    g1[0] = 0x10000;                                               // data_size=1 (2 bytes)
    g1[1] = (tile_w & 0xFFFF) << 16;                               // tensor_dim0[15:0]
    g1[2] = ((tile_w >> 16) & 0xFFFF) | ((tile_h & 0xFFFF) << 16); // dim0 hi | tensor_dim1 lo
    g1[3] = ((tile_h >> 16) & 0xFFFF) | ((tile_w & 0xFFFF) << 16); // dim1 hi | tile_dim0
    g1[4] = (tile_h & 0xFFFF);                                     // tile_dim1 (tile_dim2=0)
    g1[5] = row_stride_elems;                                      // tensor_dim0_stride[31:0]
    g1[6] = 0;
    g1[7] = 0;
    i32x4 gz = {0, 0, 0, 0};
#if __clang_major__ >= 23
    i32x8 gz8 = {0, 0, 0, 0, 0, 0, 0, 0};
    __builtin_amdgcn_tensor_load_to_lds(g0, g1, gz, gz, gz8, 0);
#else
    __builtin_amdgcn_tensor_load_to_lds(g0, g1, gz, gz, 0);
#endif
}
#endif

// ---------------------------------------------------------------------------
// f32 -> f16 convert (weights)
// ---------------------------------------------------------------------------
__global__ void k_cvt_f16(const float* __restrict__ src, _Float16* __restrict__ dst, int n) {
    int i = blockIdx.x * blockDim.x + threadIdx.x;
    if (i < n) dst[i] = (_Float16)src[i];
}

// ---------------------------------------------------------------------------
// (optional swish) -> LayerNorm over D=512 -> f16 or f32 out. One wave per row.
// ---------------------------------------------------------------------------
template <bool SWISH, bool F16OUT>
__global__ void k_swish_ln(const float* __restrict__ in,
                           const float* __restrict__ gamma,
                           const float* __restrict__ beta,
                           void* __restrict__ out) {
    const int row  = blockIdx.x * 8 + (threadIdx.x >> 5);
    const int lane = threadIdx.x & 31;
    const float* p = in + (size_t)row * Dd;

    float v[16];
    float s = 0.f, s2 = 0.f;
#pragma unroll
    for (int i = 0; i < 16; ++i) {
        float t = p[i * 32 + lane];
        if (SWISH) t = t * (1.0f / (1.0f + __expf(-t)));   // swish
        v[i] = t; s += t; s2 += t * t;
    }
#pragma unroll
    for (int o = 1; o < 32; o <<= 1) {
        s  += __shfl_xor(s,  o, 32);
        s2 += __shfl_xor(s2, o, 32);
    }
    const float mu  = s * (1.0f / Dd);
    const float var = s2 * (1.0f / Dd) - mu * mu;
    const float rs  = rsqrtf(var + EPSf);

#pragma unroll
    for (int i = 0; i < 16; ++i) {
        const int c = i * 32 + lane;
        const float o = (v[i] - mu) * rs * gamma[c] + beta[c];
        if (F16OUT) ((_Float16*)out)[(size_t)row * Dd + c] = (_Float16)o;
        else        ((float*)out)[(size_t)row * Dd + c]    = o;
    }
}

// ---------------------------------------------------------------------------
// C[m,n] = sum_k A16[m,k] * W16[n,k] (+bias). M x 512 x 512.
// One wave computes a 16x64 tile (4 accumulators) so each A fragment feeds
// 4 back-to-back WMMAs. 8 waves / block.
// ---------------------------------------------------------------------------
template <bool F16OUT, bool BIAS>
__global__ void k_gemm_wmma(const _Float16* __restrict__ A,
                            const _Float16* __restrict__ W,
                            const float*    __restrict__ bias,
                            void* __restrict__ Cout, int M) {
    const int lane = threadIdx.x & 31;
    const int wave = threadIdx.x >> 5;
    const int grp  = blockIdx.x * 8 + wave;   // (M/16)*(512/64) groups
    const int tm   = grp >> 3;                // m-tile
    const int tn0  = (grp & 7) * 4;           // first of 4 n-tiles
    if (tm * 16 >= M) return;

    const int half = lane >> 4;               // 0|1
    const int l16  = lane & 15;
    const int arow = tm * 16 + l16;

    const _Float16* Ap = A + (size_t)arow * Dd + half * 16; // A frag: K {0..15}/{16..31}
    const _Float16* Wp[4];
#pragma unroll
    for (int j = 0; j < 4; ++j)
        Wp[j] = W + (size_t)((tn0 + j) * 16 + l16) * Dd + half * 16;  // B frag: contiguous W row

    v8f acc[4];
#pragma unroll
    for (int j = 0; j < 4; ++j) acc[j] = (v8f){};

#pragma unroll 2
    for (int k = 0; k < Dd; k += 32) {
        __builtin_prefetch(Ap + k + 128, 0, 1);   // global_prefetch_b8
        HV a;
        a.h[0] = *(const v8h*)(Ap + k);
        a.h[1] = *(const v8h*)(Ap + k + 8);
#pragma unroll
        for (int j = 0; j < 4; ++j) {
            HV b;
            b.v = *(const v16h*)(Wp[j] + k);
            acc[j] = WMMA_F32_F16(a.v, b.v, acc[j]);
        }
    }

#pragma unroll
    for (int j = 0; j < 4; ++j) {
        const int bcol = (tn0 + j) * 16 + l16;
        const float bv = BIAS ? bias[bcol] : 0.0f;
#pragma unroll
        for (int r = 0; r < 8; ++r) {
            const int m = tm * 16 + half * 8 + r;
            const float o = acc[j][r] + bv;
            if (F16OUT) ((_Float16*)Cout)[(size_t)m * Dd + bcol] = (_Float16)o;
            else        ((float*)Cout)[(size_t)m * Dd + bcol]    = o;
        }
    }
}

// ---------------------------------------------------------------------------
// Flash-style masked attention. grid = (A/64, H, B), block = 128 (4 waves).
// Q/K/V are [B*A, 512] f16 (head h occupies cols h*128..h*128+127).
// Q and K tiles staged via Tensor Data Mover (async DMA) into LDS to keep
// VGPR pressure down (no scratch spills); V staged transposed so P*V
// B-fragments are contiguous 32B LDS reads. Out is [B*A, 512] f32.
// ---------------------------------------------------------------------------
__global__ void k_attn_wmma(const _Float16* __restrict__ Q,
                            const _Float16* __restrict__ K,
                            const _Float16* __restrict__ V,
                            const unsigned char* __restrict__ conn,
                            float* __restrict__ Out) {
    __shared__ __align__(32) _Float16 Qbuf[4][16 * HDd];  // 16 KB: per-wave Q tiles
    __shared__ __align__(32) _Float16 Kbuf[32 * HDd];     //  8 KB: K chunk, row-major
    __shared__ __align__(32) _Float16 VT[HDd * 32];       //  8 KB: V chunk transposed
    __shared__ __align__(32) _Float16 Pbuf[4][16 * 32];   //  4 KB: per-wave P tiles

    const int b    = blockIdx.z;
    const int h    = blockIdx.y;
    const int wave = threadIdx.x >> 5;
    const int lane = threadIdx.x & 31;
    const int half = lane >> 4;
    const int l16  = lane & 15;
    const int q0   = blockIdx.x * 64 + wave * 16;

    const size_t rowbase = (size_t)b * Aa;
    _Float16* Qw = &Qbuf[wave][0];

    // Stage this wave's 16x128 Q tile into LDS (wave-private; no block barrier).
#if USE_TDM
    tdm_load_2d_f16(Qw, Q + (rowbase + q0) * Dd + h * HDd, HDd, 16, Dd);
    __builtin_amdgcn_s_wait_tensorcnt((short)0);
#else
    {
        const _Float16* qs = Q + (rowbase + q0 + (lane >> 1)) * Dd + h * HDd + (lane & 1) * 64;
        _Float16* qd = Qw + (lane >> 1) * HDd + (lane & 1) * 64;
#pragma unroll
        for (int j = 0; j < 8; ++j)
            *(v8h*)(qd + j * 8) = *(const v8h*)(qs + j * 8);
    }
    __builtin_amdgcn_fence(__ATOMIC_ACQ_REL, "wavefront");
#endif

    v8f acc[8];
    float rmax[8], rsum[8];
#pragma unroll
    for (int t = 0; t < 8; ++t) acc[t] = (v8f){};
#pragma unroll
    for (int r = 0; r < 8; ++r) { rmax[r] = -1e30f; rsum[r] = 0.0f; }

    const int lr = threadIdx.x >> 2;        // 0..31 chunk row
    const int lc = (threadIdx.x & 3) * 32;  // col start within 128

    for (int kb = 0; kb < Aa; kb += 32) {
        __syncthreads();    // everyone done reading previous K/V chunk

        // --- K chunk: async TDM DMA of a 32x128 (rows x halfs) tile ---
#if USE_TDM
        if (wave == 0)
            tdm_load_2d_f16(&Kbuf[0], K + (rowbase + kb) * Dd + h * HDd, HDd, 32, Dd);
#else
        {
            const _Float16* ks = K + (rowbase + kb + lr) * Dd + h * HDd + lc;
#pragma unroll
            for (int j = 0; j < 4; ++j)
                *(v8h*)&Kbuf[lr * HDd + lc + j * 8] = *(const v8h*)(ks + j * 8);
        }
#endif
        // --- V chunk: cooperative transposed staging ---
        {
            const _Float16* vs = V + (rowbase + kb + lr) * Dd + h * HDd + lc;
#pragma unroll
            for (int j = 0; j < 32; ++j)
                VT[(lc + j) * 32 + lr] = vs[j];
        }
        // --- mask bytes: hoisted so all 16 loads issue as one clause ---
        unsigned char cb0[8], cb1[8];
        {
            const unsigned char* cbase = conn + ((size_t)b * Aa + q0 + half * 8) * Aa + kb;
#pragma unroll
            for (int r = 0; r < 8; ++r) {
                cb0[r] = cbase[(size_t)r * Aa + l16];
                cb1[r] = cbase[(size_t)r * Aa + 16 + l16];
            }
        }
#if USE_TDM
        if (wave == 0) __builtin_amdgcn_s_wait_tensorcnt((short)0);
#endif
        __syncthreads();

        // S = Q * K^T for 16 queries x 32 keys (two 16-wide column tiles)
        v8f s0 = {}, s1 = {};
#pragma unroll
        for (int c = 0; c < 4; ++c) {
            HV a, bk0, bk1;
            a.h[0] = *(const v8h*)&Qw[l16 * HDd + c * 32 + half * 16];
            a.h[1] = *(const v8h*)&Qw[l16 * HDd + c * 32 + half * 16 + 8];
            bk0.v  = *(const v16h*)&Kbuf[l16 * HDd + c * 32 + half * 16];
            bk1.v  = *(const v16h*)&Kbuf[(16 + l16) * HDd + c * 32 + half * 16];
            s0 = WMMA_F32_F16(a.v, bk0.v, s0);
            s1 = WMMA_F32_F16(a.v, bk1.v, s1);
        }

        // mask + online softmax (rows split across lane halves per C layout)
        _Float16* P = &Pbuf[wave][0];
#pragma unroll
        for (int r = 0; r < 8; ++r) {
            float e0 = cb0[r] ? s0[r] * SCALEf : -1e30f;
            float e1 = cb1[r] ? s1[r] * SCALEf : -1e30f;

            float m = fmaxf(e0, e1);
#pragma unroll
            for (int o = 1; o < 16; o <<= 1) m = fmaxf(m, __shfl_xor(m, o, 32));
            const float nm   = fmaxf(rmax[r], m);
            const float corr = __expf(rmax[r] - nm);
            rmax[r] = nm;

            const float p0 = __expf(e0 - nm);
            const float p1 = __expf(e1 - nm);
            float ts = p0 + p1;
#pragma unroll
            for (int o = 1; o < 16; o <<= 1) ts += __shfl_xor(ts, o, 32);
            rsum[r] = rsum[r] * corr + ts;

#pragma unroll
            for (int t = 0; t < 8; ++t) acc[t][r] *= corr;

            P[(half * 8 + r) * 32 + l16]      = (_Float16)p0;
            P[(half * 8 + r) * 32 + 16 + l16] = (_Float16)p1;
        }
        // Pbuf is wave-private; DS ops of one wave complete in order, so a
        // wavefront fence (compiler reorder barrier) is enough — no barrier.
        __builtin_amdgcn_fence(__ATOMIC_ACQ_REL, "wavefront");

        // O += P(16x32) * V(32x128) : 8 column tiles
        HV pa;
        pa.h[0] = *(const v8h*)&P[l16 * 32 + half * 16];
        pa.h[1] = *(const v8h*)&P[l16 * 32 + 8 + half * 16];
#pragma unroll
        for (int t = 0; t < 8; ++t) {
            HV bv;
            bv.v = *(const v16h*)&VT[(t * 16 + l16) * 32 + half * 16];
            acc[t] = WMMA_F32_F16(pa.v, bv.v, acc[t]);
        }
    }

    // normalize and store
#pragma unroll
    for (int t = 0; t < 8; ++t) {
#pragma unroll
        for (int r = 0; r < 8; ++r) {
            const int qrow = q0 + half * 8 + r;
            Out[(rowbase + qrow) * Dd + h * HDd + t * 16 + l16] = acc[t][r] / rsum[r];
        }
    }
}

// ---------------------------------------------------------------------------
// Host-side orchestration
// ---------------------------------------------------------------------------
extern "C" void kernel_launch(void* const* d_in, const int* in_sizes, int n_in,
                              void* d_out, int out_size, void* d_ws, size_t ws_size,
                              hipStream_t stream) {
    const float* x       = (const float*)d_in[0];
    const unsigned char* conn = (const unsigned char*)d_in[1];
    const float* Wq      = (const float*)d_in[2];
    const float* Wk      = (const float*)d_in[3];
    const float* Wv      = (const float*)d_in[4];
    const float* norm_w  = (const float*)d_in[5];
    const float* norm_b  = (const float*)d_in[6];
    const float* ln1_w   = (const float*)d_in[7];
    const float* ln1_b   = (const float*)d_in[8];
    const float* fc1_w   = (const float*)d_in[9];
    const float* fc1_b   = (const float*)d_in[10];
    const float* ln2_w   = (const float*)d_in[11];
    const float* ln2_b   = (const float*)d_in[12];
    const float* fc2_w   = (const float*)d_in[13];
    const float* fc2_b   = (const float*)d_in[14];
    float* out           = (float*)d_out;

    // workspace layout (all offsets 256B aligned)
    size_t off = 0;
    auto alloc = [&](size_t bytes) { size_t o = off; off += (bytes + 255) & ~(size_t)255; return o; };
    char* ws = (char*)d_ws;
    const size_t MD16 = (size_t)Mm * Dd * sizeof(_Float16);   // 8 MB
    const size_t MD32 = (size_t)Mm * Dd * sizeof(float);      // 16 MB
    const size_t WD16 = (size_t)Dd * Dd * sizeof(_Float16);   // 0.5 MB

    _Float16* X16  = (_Float16*)(ws + alloc(MD16));
    _Float16* Wq16 = (_Float16*)(ws + alloc(WD16));
    _Float16* Wk16 = (_Float16*)(ws + alloc(WD16));
    _Float16* Wv16 = (_Float16*)(ws + alloc(WD16));
    _Float16* W116 = (_Float16*)(ws + alloc(WD16));
    _Float16* W216 = (_Float16*)(ws + alloc(WD16));
    _Float16* Q16  = (_Float16*)(ws + alloc(MD16));
    _Float16* K16  = (_Float16*)(ws + alloc(MD16));
    _Float16* V16  = (_Float16*)(ws + alloc(MD16));
    float*    F0   = (float*)(ws + alloc(MD32));              // attn out / fc2 out
    float*    F1   = (float*)(ws + alloc(MD32));              // fc1 out
    _Float16* G16  = (_Float16*)(ws + alloc(MD16));           // post swish+LN f16

    const int nW = Dd * Dd;
    // 1) weights -> f16
    k_cvt_f16<<<(nW + 255) / 256, 256, 0, stream>>>(Wq,    Wq16, nW);
    k_cvt_f16<<<(nW + 255) / 256, 256, 0, stream>>>(Wk,    Wk16, nW);
    k_cvt_f16<<<(nW + 255) / 256, 256, 0, stream>>>(Wv,    Wv16, nW);
    k_cvt_f16<<<(nW + 255) / 256, 256, 0, stream>>>(fc1_w, W116, nW);
    k_cvt_f16<<<(nW + 255) / 256, 256, 0, stream>>>(fc2_w, W216, nW);

    const int lnBlocks = Mm / 8;                      // 1024
    const int gmBlocks = (Mm / 16) * (Dd / 64) / 8;   // 512

    // 2) xn = LN(x) -> f16
    k_swish_ln<false, true><<<lnBlocks, 256, 0, stream>>>(x, norm_w, norm_b, X16);

    // 3) Q/K/V projections (f16 out, no bias)
    k_gemm_wmma<true, false><<<gmBlocks, 256, 0, stream>>>(X16, Wq16, nullptr, Q16, Mm);
    k_gemm_wmma<true, false><<<gmBlocks, 256, 0, stream>>>(X16, Wk16, nullptr, K16, Mm);
    k_gemm_wmma<true, false><<<gmBlocks, 256, 0, stream>>>(X16, Wv16, nullptr, V16, Mm);

    // 4) masked flash attention
    dim3 agrid(Aa / 64, Hh, Bb);
    k_attn_wmma<<<agrid, 128, 0, stream>>>(Q16, K16, V16, conn, F0);

    // 5) h = LN(swish(attn)) @ fc1^T + b1
    k_swish_ln<true, true><<<lnBlocks, 256, 0, stream>>>(F0, ln1_w, ln1_b, G16);
    k_gemm_wmma<false, true><<<gmBlocks, 256, 0, stream>>>(G16, W116, fc1_b, F1, Mm);

    // 6) h2 = LN(swish(h)) @ fc2^T + b2
    k_swish_ln<true, true><<<lnBlocks, 256, 0, stream>>>(F1, ln2_w, ln2_b, G16);
    k_gemm_wmma<false, true><<<gmBlocks, 256, 0, stream>>>(G16, W216, fc2_b, F0, Mm);

    // 7) final LN -> d_out (f32)
    k_swish_ln<false, false><<<lnBlocks, 256, 0, stream>>>(F0, norm_w, norm_b, out);
}
